// Encoder_1151051235706
// MI455X (gfx1250) — compile-verified, matrix operand order checked by reading, not run on hardware
//
#include <hip/hip_runtime.h>

typedef __attribute__((ext_vector_type(16))) __bf16 v16bf;
typedef __attribute__((ext_vector_type(8)))  float  v8f;

#define E_SZ 256
#define U_SZ 512
#define B_SZ 64
#define T_SZ 256
#define G3   1536                 // 3*U
#define MROWS (B_SZ * T_SZ)       // 16384
#define NWG  32                   // scan workgroups
#define SLICE (U_SZ / NWG)        // 16 hidden columns per WG
#define LDS_ELEMS (3 * 16 * 32 * 16)   // 24576 bf16 = 48 KB
#define NT_CNT (G3 / 16)          // 96 N-tiles
#define NTG    4                  // N-tiles per wave in proj
#define WF_FRAGS (NT_CNT * (E_SZ / 32) * 32)   // 96*8*32 = 24576 fragments

union Frag16 {
    v16bf v;
    uint4 q[2];
    unsigned short s[16];
};

__device__ __forceinline__ unsigned short f2bf(float f) {
    unsigned int u = __float_as_uint(f);
    u += 0x7FFFu + ((u >> 16) & 1u);       // round-to-nearest-even
    return (unsigned short)(u >> 16);
}
__device__ __forceinline__ float bf2f(unsigned short s) {
    return __uint_as_float(((unsigned int)s) << 16);
}
__device__ __forceinline__ float sigmoidf_(float v) {
    return 1.0f / (1.0f + __expf(-v));
}

// ---------------------------------------------------------------------------
// Kernel 0: zero barrier counters; convert initial hidden -> bf16
// ---------------------------------------------------------------------------
__global__ void enc_init(const float* __restrict__ hidden,
                         unsigned short* __restrict__ h0,
                         int* __restrict__ cnt) {
    int i = blockIdx.x * blockDim.x + threadIdx.x;
    if (i < B_SZ * U_SZ) h0[i] = f2bf(hidden[i]);
    if (i < T_SZ)        cnt[i] = 0;
}

// ---------------------------------------------------------------------------
// Kernel 1a: gather + convert embedding rows once: xe_bf[row][256] (bf16).
// ---------------------------------------------------------------------------
__global__ void enc_prep_emb(const int* __restrict__ x,
                             const float* __restrict__ emb,
                             unsigned short* __restrict__ xe_bf) {
    const int i   = blockIdx.x * blockDim.x + threadIdx.x;  // 0 .. 16384*32-1
    const int row = i >> 5;
    const int seg = (i & 31) * 8;
    const int vidx = x[row];
    const float4* p = (const float4*)(emb + (long)vidx * E_SZ + seg);
    const float4 a = p[0], b = p[1];
    Frag16 o;
    o.s[0] = f2bf(a.x); o.s[1] = f2bf(a.y); o.s[2] = f2bf(a.z); o.s[3] = f2bf(a.w);
    o.s[4] = f2bf(b.x); o.s[5] = f2bf(b.y); o.s[6] = f2bf(b.z); o.s[7] = f2bf(b.w);
    *(uint4*)(xe_bf + (long)row * E_SZ + seg) = o.q[0];
}

// ---------------------------------------------------------------------------
// Kernel 1b: W -> bf16, pre-swizzled in B-fragment layout:
//   wf[((nt*8 + ks)*32 + lane)*16 + j],  k = ks*32 + 16*(j/8) + 8*(lane/16) + j%8
// ---------------------------------------------------------------------------
__global__ void enc_prep_w(const float* __restrict__ W,
                           unsigned short* __restrict__ wf) {
    const int f = blockIdx.x * blockDim.x + threadIdx.x;    // 0 .. 24575
    const int lane = f & 31;
    const int ks   = (f >> 5) & 7;
    const int nt   = f >> 8;
    const int n    = nt * 16 + (lane & 15);
    Frag16 o;
#pragma unroll
    for (int j = 0; j < 16; ++j) {
        const int k = ks * 32 + (j >> 3) * 16 + (lane >> 4) * 8 + (j & 7);
        o.s[j] = f2bf(W[k * G3 + n]);
    }
    uint4* d = (uint4*)(wf + (long)f * 16);
    d[0] = o.q[0];
    d[1] = o.q[1];
}

// ---------------------------------------------------------------------------
// Kernel 2: xp[t][b][:] = xe_bf[b*T+t] @ W + b0  (16384 x 256 x 1536)
// 8 waves/block; each wave: 1 M-tile x 4 N-tiles. All fragment addressing is
// loop-invariant base + immediate offset; note xp is stored [T][B][3U] so the
// scan reads it with tiny strides.
// ---------------------------------------------------------------------------
__global__ void enc_proj(const unsigned short* __restrict__ xe_bf,
                         const unsigned short* __restrict__ wf,
                         const float* __restrict__ bias0,
                         float* __restrict__ xp) {
    const int lane = threadIdx.x & 31;
    const int wave = threadIdx.x >> 5;
    const int lr   = lane & 15;
    const int half = lane >> 4;

    const int job = blockIdx.x * 8 + wave;      // 0 .. 1024*24-1
    const int ntg = job % (NT_CNT / NTG);       // 0..23
    const int mt  = job / (NT_CNT / NTG);       // 0..1023

    // loop-invariant bases
    const unsigned short* ap = xe_bf + (size_t)(mt * 16 + lr) * E_SZ + half * 8;
    const unsigned short* wb = wf + ((size_t)(ntg * NTG) * 8 * 32 + lane) * 16;

    const v8f z8 = {0.f, 0.f, 0.f, 0.f, 0.f, 0.f, 0.f, 0.f};
    v8f acc[NTG] = {z8, z8, z8, z8};
#pragma unroll
    for (int ks = 0; ks < E_SZ / 32; ++ks) {
        Frag16 a;
        a.q[0] = *(const uint4*)(ap + ks * 32);
        a.q[1] = *(const uint4*)(ap + ks * 32 + 16);
#pragma unroll
        for (int c = 0; c < NTG; ++c) {
            const uint4* bp = (const uint4*)(wb + (size_t)(c * 8 + ks) * 512);
            Frag16 bm;
            bm.q[0] = bp[0];
            bm.q[1] = bp[1];
            acc[c] = __builtin_amdgcn_wmma_f32_16x16x32_bf16(
                false, a.v, false, bm.v, (short)0, acc[c], false, false);
        }
    }
    // store: row r = mt*16 + half*8 + i  ->  b = r/256, t = r%256.
    // 16-row tile never crosses a batch boundary, so b is tile-constant.
    const int bidx = mt >> 4;
    const int t0   = (mt & 15) * 16 + half * 8;
#pragma unroll
    for (int c = 0; c < NTG; ++c) {
        const int ncol = (ntg * NTG + c) * 16 + lr;
        const float b0 = bias0[ncol];
        float* op = xp + ((size_t)t0 * B_SZ + bidx) * G3 + ncol;
#pragma unroll
        for (int i = 0; i < 8; ++i) {
            op[(size_t)i * B_SZ * G3] = acc[c][i] + b0;   // t0+i, stride 384 KB
        }
    }
}

// ---------------------------------------------------------------------------
// Kernel 3: persistent GRU scan. 32 WGs x 128 threads (4 waves = 4 M-tiles).
// Uw slice lives in LDS (bf16, fragment layout) for all 256 steps.
// ---------------------------------------------------------------------------
__global__ void enc_scan(const int* __restrict__ x,
                         const float* __restrict__ xp,   // [T][B][3U]
                         const float* __restrict__ Uw,
                         const float* __restrict__ bias, // [2,1536]
                         unsigned short* __restrict__ h0,
                         unsigned short* __restrict__ h1,
                         int* __restrict__ cnt,
                         float* __restrict__ out) {
    extern __shared__ unsigned short lds[];
    const int lane  = threadIdx.x & 31;
    const int w     = threadIdx.x >> 5;            // M-tile 0..3
    const int lr    = lane & 15;
    const int half  = lane >> 4;
    const int sbase = blockIdx.x * SLICE;          // hidden column base
    const int hcol  = sbase + lr;

    // Stage Uw slice into LDS, fragment-major: [gate][kstep][lane][16 bf16]
    for (int e = threadIdx.x; e < LDS_ELEMS; e += blockDim.x) {
        const int j  = e & 15;
        const int ln = (e >> 4) & 31;
        const int ks = (e >> 9) & 15;
        const int g  = e >> 13;
        const int k  = ks * 32 + (j >> 3) * 16 + (ln >> 4) * 8 + (j & 7);
        const int c  = g * U_SZ + sbase + (ln & 15);
        lds[e] = f2bf(Uw[k * G3 + c]);
    }
    __syncthreads();

    const float b1z = bias[G3 + hcol];
    const float b1r = bias[G3 + U_SZ + hcol];
    const float b1h = bias[G3 + 2 * U_SZ + hcol];

    unsigned short* hc = h0;
    unsigned short* hn = h1;
    float* stateOut = out + (size_t)B_SZ * T_SZ * U_SZ;
    const v8f z8 = {0.f, 0.f, 0.f, 0.f, 0.f, 0.f, 0.f, 0.f};

    const int rowT = w * 16 + half * 8;            // first batch row of C rows
    const unsigned short* lb = lds + lane * 16;    // LDS fragment base

    for (int t = 0; t < T_SZ; ++t) {
        // --- prefetch step-local scalars (all base + immediate) ---
        const float* xpb = xp + ((size_t)t * B_SZ + rowT) * G3 + hcol;
        const unsigned short* hob = hc + (size_t)rowT * U_SZ + hcol;
        const int* xb = x + (size_t)rowT * T_SZ + t;
        float xz[8], xr[8], xh[8], ho[8];
        int   msk[8];
#pragma unroll
        for (int i = 0; i < 8; ++i) {
            xz[i]  = xpb[(size_t)i * G3];
            xr[i]  = xpb[(size_t)i * G3 + U_SZ];
            xh[i]  = xpb[(size_t)i * G3 + 2 * U_SZ];
            ho[i]  = bf2f(hob[(size_t)i * U_SZ]);
            msk[i] = xb[(size_t)i * T_SZ];
        }

        v8f a0 = z8, a1 = z8, a2 = z8;
        const unsigned short* hp = hc + (size_t)(w * 16 + lr) * U_SZ + half * 8;
#pragma unroll
        for (int ks = 0; ks < 16; ++ks) {
            Frag16 a, bz, br, bh;
            a.q[0] = *(const uint4*)(hp + ks * 32);
            a.q[1] = *(const uint4*)(hp + ks * 32 + 16);
            const uint4* p;
            p = (const uint4*)(lb + (size_t)(0 * 16 + ks) * 512);
            bz.q[0] = p[0]; bz.q[1] = p[1];
            p = (const uint4*)(lb + (size_t)(1 * 16 + ks) * 512);
            br.q[0] = p[0]; br.q[1] = p[1];
            p = (const uint4*)(lb + (size_t)(2 * 16 + ks) * 512);
            bh.q[0] = p[0]; bh.q[1] = p[1];
            a0 = __builtin_amdgcn_wmma_f32_16x16x32_bf16(
                false, a.v, false, bz.v, (short)0, a0, false, false);
            a1 = __builtin_amdgcn_wmma_f32_16x16x32_bf16(
                false, a.v, false, br.v, (short)0, a1, false, false);
            a2 = __builtin_amdgcn_wmma_f32_16x16x32_bf16(
                false, a.v, false, bh.v, (short)0, a2, false, false);
        }

        unsigned short* hnb = hn + (size_t)rowT * U_SZ + hcol;
        float* ob = out + ((size_t)rowT * T_SZ + t) * U_SZ + hcol;
        float* sb = stateOut + (size_t)rowT * U_SZ + hcol;
#pragma unroll
        for (int i = 0; i < 8; ++i) {
            const float zg = sigmoidf_(xz[i] + a0[i] + b1z);
            const float rg = sigmoidf_(xr[i] + a1[i] + b1r);
            const float cd = tanhf(xh[i] + rg * (a2[i] + b1h));
            float hv = zg * ho[i] + (1.0f - zg) * cd;
            if (msk[i] == 0) hv = ho[i];               // mask_zero carry
            hnb[(size_t)i * U_SZ] = f2bf(hv);
            ob[(size_t)i * T_SZ * U_SZ] = hv;
            if (t == T_SZ - 1) sb[(size_t)i * U_SZ] = hv;
        }

        // device-wide step barrier (all 32 WGs resident)
        __threadfence();
        __syncthreads();
        if (threadIdx.x == 0) {
            atomicAdd(&cnt[t], 1);
            while (__hip_atomic_load(&cnt[t], __ATOMIC_ACQUIRE,
                                     __HIP_MEMORY_SCOPE_AGENT) < NWG) {
                __builtin_amdgcn_s_sleep(1);
            }
        }
        __syncthreads();
        __threadfence();

        unsigned short* tmp = hc; hc = hn; hn = tmp;   // swap buffers
    }
}

// ---------------------------------------------------------------------------
extern "C" void kernel_launch(void* const* d_in, const int* in_sizes, int n_in,
                              void* d_out, int out_size, void* d_ws, size_t ws_size,
                              hipStream_t stream) {
    const int*   x      = (const int*)d_in[0];
    const float* hidden = (const float*)d_in[1];
    const float* emb    = (const float*)d_in[2];
    const float* W      = (const float*)d_in[3];
    const float* Uw     = (const float*)d_in[4];
    const float* bias   = (const float*)d_in[5];
    float* out = (float*)d_out;

    char* ws = (char*)d_ws;
    size_t off = 0;
    float* xp = (float*)(ws + off);               off += (size_t)MROWS * G3 * 4;   // 100.7 MB
    unsigned short* xe_bf = (unsigned short*)(ws + off); off += (size_t)MROWS * E_SZ * 2;  // 8 MB
    unsigned short* wf    = (unsigned short*)(ws + off); off += (size_t)WF_FRAGS * 16 * 2; // 768 KB
    unsigned short* h0 = (unsigned short*)(ws + off);    off += (size_t)B_SZ * U_SZ * 2;
    unsigned short* h1 = (unsigned short*)(ws + off);    off += (size_t)B_SZ * U_SZ * 2;
    int* cnt = (int*)(ws + off);

    enc_init<<<(B_SZ * U_SZ + 255) / 256, 256, 0, stream>>>(hidden, h0, cnt);
    enc_prep_emb<<<(MROWS * (E_SZ / 8)) / 256, 256, 0, stream>>>(x, emb, xe_bf);
    enc_prep_w<<<WF_FRAGS / 256, 256, 0, stream>>>(W, wf);
    enc_proj<<<(MROWS / 16) * (NT_CNT / NTG) / 8, 256, 0, stream>>>(xe_bf, wf, bias, xp);
    enc_scan<<<NWG, 128, LDS_ELEMS * sizeof(unsigned short), stream>>>(
        x, xp, Uw, bias, h0, h1, cnt, out);
}
